// MLP_Model_42073499631624
// MI455X (gfx1250) — matrix operand order repaired and checked by reading, not compiled, not run on hardware
//
#include <hip/hip_runtime.h>
#include <hip/hip_bf16.h>

// ---------------------------------------------------------------------------
// CDNA5 (gfx1250, wave32). All GEMMs via v_wmma_f32_16x16x32_bf16.
// Weights pre-packed to B-fragment layout (bf16, 2x b128 per fragment);
// GRU state: f32 carry + bf16 shadow (A operand, 2x b128 per fragment).
// Branch-free zero-padded A loaders; fast exp/rcp-based tanh & sigmoid.
// ---------------------------------------------------------------------------

typedef __bf16 bf16_t;
typedef bf16_t v16bf __attribute__((ext_vector_type(16)));
typedef float  v8f   __attribute__((ext_vector_type(8)));

#define FEAT   36
#define NNODE  24
#define NROUTE 48
#define BATCH  512
#define NSEQ   (BATCH * NROUTE)   /* 24576 */
#define NTOK   (NSEQ * NNODE)     /* 589824 */
#define CH     128
#define CO     32
#define RH     128
#define G3     384

static __device__ __forceinline__ v8f zero8() {
    v8f z;
#pragma unroll
    for (int i = 0; i < 8; ++i) z[i] = 0.0f;
    return z;
}

// fast activations: v_exp_f32 + v_rcp_f32 (saturate correctly at the rails)
static __device__ __forceinline__ float fast_tanhf(float x) {
    float e = __expf(2.0f * x);
    return 1.0f - 2.0f * __builtin_amdgcn_rcpf(e + 1.0f);
}
static __device__ __forceinline__ float fast_sigmoidf(float x) {
    return __builtin_amdgcn_rcpf(1.0f + __expf(-x));
}

// ---- A operand (16x32) from row-major f32, fully in-bounds k-tile:
//      branch-free, 4x float4 loads + packed cvt ------------------------------
static __device__ __forceinline__ v16bf
load_a_f32_full(const float* __restrict__ A, int lda, int row0, int kt, int lane) {
    const int half = (lane >> 4) & 1;
    const float4* Aq = (const float4*)(A + (long)(row0 + (lane & 15)) * lda);
    const int f0 = (kt + half * 8) >> 2;      // float4 index of K-group 0
    float4 g0 = Aq[f0], g1 = Aq[f0 + 1];      // kk0 .. kk0+7
    float4 g2 = Aq[f0 + 4], g3 = Aq[f0 + 5];  // kk0+16 .. kk0+23
    v16bf a;
    a[0] = (bf16_t)g0.x; a[1] = (bf16_t)g0.y; a[2] = (bf16_t)g0.z; a[3] = (bf16_t)g0.w;
    a[4] = (bf16_t)g1.x; a[5] = (bf16_t)g1.y; a[6] = (bf16_t)g1.z; a[7] = (bf16_t)g1.w;
    a[8] = (bf16_t)g2.x; a[9] = (bf16_t)g2.y; a[10] = (bf16_t)g2.z; a[11] = (bf16_t)g2.w;
    a[12] = (bf16_t)g3.x; a[13] = (bf16_t)g3.y; a[14] = (bf16_t)g3.z; a[15] = (bf16_t)g3.w;
    return a;
}

// ---- tail k-tile for K=36 (valid kk 32..35 -> lanes half==0, a[0..3]) -------
static __device__ __forceinline__ v16bf
load_a_f32_tail36(const float* __restrict__ A, int lda, int row0, int lane) {
    const bool hi = ((lane >> 4) & 1) != 0;
    const float* Ar = A + (long)(row0 + (lane & 15)) * lda;
    float4 g = *(const float4*)(Ar + 32);     // elements 32..35, always in-bounds
    const bf16_t z = (bf16_t)0.0f;
    v16bf a;
#pragma unroll
    for (int i = 4; i < 16; ++i) a[i] = z;
    a[0] = hi ? z : (bf16_t)g.x;
    a[1] = hi ? z : (bf16_t)g.y;
    a[2] = hi ? z : (bf16_t)g.z;
    a[3] = hi ? z : (bf16_t)g.w;
    return a;
}

// ---- tail k-tile for K=72, kt=64 (valid kk 64..71 -> half==0, a[0..7]) ------
static __device__ __forceinline__ v16bf
load_a_f32_tail72(const float* __restrict__ A, int lda, int row0, int lane) {
    const bool hi = ((lane >> 4) & 1) != 0;
    const float* Ar = A + (long)(row0 + (lane & 15)) * lda;
    float4 g0 = *(const float4*)(Ar + 64);    // 64..67
    float4 g1 = *(const float4*)(Ar + 68);    // 68..71
    const bf16_t z = (bf16_t)0.0f;
    v16bf a;
#pragma unroll
    for (int i = 8; i < 16; ++i) a[i] = z;
    a[0] = hi ? z : (bf16_t)g0.x;
    a[1] = hi ? z : (bf16_t)g0.y;
    a[2] = hi ? z : (bf16_t)g0.z;
    a[3] = hi ? z : (bf16_t)g0.w;
    a[4] = hi ? z : (bf16_t)g1.x;
    a[5] = hi ? z : (bf16_t)g1.y;
    a[6] = hi ? z : (bf16_t)g1.z;
    a[7] = hi ? z : (bf16_t)g1.w;
    return a;
}

// ---- A operand from row-major bf16 (global or LDS), two b128 loads ----------
static __device__ __forceinline__ v16bf
load_a_bf16(const bf16_t* __restrict__ A, int lda, int row0, int kt, int lane) {
    const int half = (lane >> 4) & 1;
    const uint4* Aq = (const uint4*)(A + (long)(row0 + (lane & 15)) * lda);
    const int q0 = (kt >> 3) + half;
    union { v16bf v; uint4 q[2]; } r;
    r.q[0] = Aq[q0];
    r.q[1] = Aq[q0 + 2];
    return r.v;
}

// ---- B operand from pre-packed fragment buffer: 32B contiguous per lane -----
static __device__ __forceinline__ v16bf
load_b_packed(const bf16_t* __restrict__ P, int ntiles, int ktile, int ntile, int lane) {
    const bf16_t* p = P + (((long)ktile * ntiles + ntile) * 32 + lane) * 16;
    return *(const v16bf*)p;
}

#define WMMA_BF16(a, b, c) \
    __builtin_amdgcn_wmma_f32_16x16x32_bf16(false, (a), false, (b), (short)0, (c), false, false)

// ---------------------------------------------------------------------------
// Pack f32 [K,N] weights into bf16 B-fragment layout, zero-fill past K.
// ---------------------------------------------------------------------------
__global__ void pack_b_kernel(const float* __restrict__ B, bf16_t* __restrict__ out,
                              int K, int N, int ktiles, int ntiles) {
    const int idx = blockIdx.x * blockDim.x + threadIdx.x;
    const int total = ktiles * ntiles * 32;
    if (idx >= total) return;
    const int lane = idx & 31;
    const int tile = idx >> 5;
    const int nt = tile % ntiles, kt = tile / ntiles;
    const int n = nt * 16 + (lane & 15);
    const int kbase = kt * 32 + ((lane >> 4) ? 8 : 0);
    bf16_t* o = out + (long)idx * 16;
#pragma unroll
    for (int v = 0; v < 8; ++v) {
        int kk = kbase + 2 * v + ((v >= 4) ? 8 : 0);
        o[2 * v]     = (bf16_t)((kk     < K) ? B[(long)kk * N + n]       : 0.0f);
        o[2 * v + 1] = (bf16_t)((kk + 1 < K) ? B[(long)(kk + 1) * N + n] : 0.0f);
    }
}

// ---------------------------------------------------------------------------
// Kernel 1: fused customer MLP  tanh(tanh(X @ W1 + b1) @ W2 + b2)
// ---------------------------------------------------------------------------
__global__ void __launch_bounds__(128) customer_mlp_kernel(
    const float* __restrict__ cust,   // [NSEQ, NNODE*FEAT]
    const bf16_t* __restrict__ w1p,   // packed: ktiles=2, ntiles=8
    const float* __restrict__ b1,
    const bf16_t* __restrict__ w2p,   // packed: ktiles=4, ntiles=2
    const float* __restrict__ b2,
    bf16_t* __restrict__ h2out)       // [NTOK, CO], token m = t*NSEQ + n
{
    __shared__ __align__(16) bf16_t smem[4][16][CH];
    const int lane  = threadIdx.x & 31;
    const int wv    = threadIdx.x >> 5;
    const int strip = blockIdx.x * 4 + wv;
    const int m0    = strip * 16;
    const int t     = m0 / NSEQ;
    const int n0    = m0 % NSEQ;

    v8f acc[8];
#pragma unroll
    for (int j = 0; j < 8; ++j) acc[j] = zero8();

    const float* Arow = cust + (long)t * FEAT;
    {
        v16bf a = load_a_f32_full(Arow, NNODE * FEAT, n0, 0, lane);
#pragma unroll
        for (int j = 0; j < 8; ++j)
            acc[j] = WMMA_BF16(a, load_b_packed(w1p, 8, 0, j, lane), acc[j]);
    }
    {
        v16bf a = load_a_f32_tail36(Arow, NNODE * FEAT, n0, lane);
#pragma unroll
        for (int j = 0; j < 8; ++j)
            acc[j] = WMMA_BF16(a, load_b_packed(w1p, 8, 1, j, lane), acc[j]);
    }

    const int cl = lane & 15, hf = (lane >> 4) & 1;
#pragma unroll
    for (int j = 0; j < 8; ++j) {
        const int col = j * 16 + cl;
        const float bb = b1[col];
#pragma unroll
        for (int r = 0; r < 8; ++r)
            smem[wv][r + hf * 8][col] = (bf16_t)fast_tanhf(acc[j][r] + bb);
    }
    __syncthreads();

    v8f acc2[2];
    acc2[0] = zero8(); acc2[1] = zero8();
#pragma unroll
    for (int ki = 0; ki < 4; ++ki) {
        v16bf a = load_a_bf16(&smem[wv][0][0], CH, 0, ki * 32, lane);
#pragma unroll
        for (int j = 0; j < 2; ++j)
            acc2[j] = WMMA_BF16(a, load_b_packed(w2p, 2, ki, j, lane), acc2[j]);
    }
#pragma unroll
    for (int j = 0; j < 2; ++j) {
        const int col = j * 16 + cl;
        const float bb = b2[col];
#pragma unroll
        for (int r = 0; r < 8; ++r) {
            const long m = m0 + r + hf * 8;
            h2out[m * CO + col] = (bf16_t)fast_tanhf(acc2[j][r] + bb);
        }
    }
}

// ---------------------------------------------------------------------------
// Kernel 2: one GRU timestep, M-blocked x2 (32x16 output tile per wave).
// ---------------------------------------------------------------------------
template <int KIN>
__global__ void __launch_bounds__(128) gru_step_kernel(
    const float*  __restrict__ Hin,    // [NSEQ, RH] f32 carry (read)
    float*        __restrict__ Hout,   // [NSEQ, RH] f32 carry (write)
    const bf16_t* __restrict__ Hbf_in, // [NSEQ, RH] bf16 shadow (A operand)
    bf16_t*       __restrict__ Hbf_out,// [NSEQ, RH] bf16 shadow (write)
    const bf16_t* __restrict__ X,      // [NSEQ, KIN] bf16 input
    const bf16_t* __restrict__ wihp,   // packed, ktiles=KIN/32, ntiles=24
    const bf16_t* __restrict__ whhp,   // packed, ktiles=4,      ntiles=24
    const float* __restrict__ bih, const float* __restrict__ bhh)
{
    const int lane = threadIdx.x & 31;
    const int wg = blockIdx.x * 4 + (threadIdx.x >> 5);  // (NSEQ/32)*8 waves
    const int mt = wg >> 3, nt = wg & 7;
    const int m0 = mt * 32, n0 = nt * 16;

    v8f ar[2], az[2], ain[2], ahn[2];
#pragma unroll
    for (int i = 0; i < 2; ++i) { ar[i] = zero8(); az[i] = zero8(); ain[i] = zero8(); ahn[i] = zero8(); }

    // hidden: h @ whh  (K = 128)
#pragma unroll
    for (int ki = 0; ki < 4; ++ki) {
        v16bf a0 = load_a_bf16(Hbf_in, RH, m0,      ki * 32, lane);
        v16bf a1 = load_a_bf16(Hbf_in, RH, m0 + 16, ki * 32, lane);
        v16bf br = load_b_packed(whhp, 24, ki, nt,      lane);
        ar[0] = WMMA_BF16(a0, br, ar[0]);  ar[1] = WMMA_BF16(a1, br, ar[1]);
        v16bf bz = load_b_packed(whhp, 24, ki, nt + 8,  lane);
        az[0] = WMMA_BF16(a0, bz, az[0]);  az[1] = WMMA_BF16(a1, bz, az[1]);
        v16bf bn = load_b_packed(whhp, 24, ki, nt + 16, lane);
        ahn[0] = WMMA_BF16(a0, bn, ahn[0]); ahn[1] = WMMA_BF16(a1, bn, ahn[1]);
    }
    // input: x @ wih  (K = KIN)
#pragma unroll
    for (int ki = 0; ki < KIN / 32; ++ki) {
        v16bf a0 = load_a_bf16(X, KIN, m0,      ki * 32, lane);
        v16bf a1 = load_a_bf16(X, KIN, m0 + 16, ki * 32, lane);
        v16bf br = load_b_packed(wihp, 24, ki, nt,      lane);
        ar[0] = WMMA_BF16(a0, br, ar[0]);  ar[1] = WMMA_BF16(a1, br, ar[1]);
        v16bf bz = load_b_packed(wihp, 24, ki, nt + 8,  lane);
        az[0] = WMMA_BF16(a0, bz, az[0]);  az[1] = WMMA_BF16(a1, bz, az[1]);
        v16bf bn = load_b_packed(wihp, 24, ki, nt + 16, lane);
        ain[0] = WMMA_BF16(a0, bn, ain[0]); ain[1] = WMMA_BF16(a1, bn, ain[1]);
    }

    const int cl = lane & 15, hf = (lane >> 4) & 1;
    const int nc = n0 + cl;
    const float br_  = bih[nc]       + bhh[nc];
    const float bz_  = bih[nc + 128] + bhh[nc + 128];
    const float bin_ = bih[nc + 256];
    const float bhn_ = bhh[nc + 256];
#pragma unroll
    for (int i = 0; i < 2; ++i) {
#pragma unroll
        for (int r = 0; r < 8; ++r) {
            const long m = m0 + i * 16 + r + hf * 8;
            const float hold = Hin[m * RH + nc];
            const float rg = fast_sigmoidf(ar[i][r] + br_);
            const float zg = fast_sigmoidf(az[i][r] + bz_);
            const float ng = fast_tanhf(ain[i][r] + bin_ + rg * (ahn[i][r] + bhn_));
            const float hn = (1.0f - zg) * ng + zg * hold;
            Hout[m * RH + nc] = hn;
            Hbf_out[m * RH + nc] = (bf16_t)hn;
        }
    }
}

// ---------------------------------------------------------------------------
// Kernel 3: mean over 48 routes -> Hcat[:, 0:128] (bf16, ldc=192)
// ---------------------------------------------------------------------------
__global__ void route_mean_kernel(const float* __restrict__ Hlast,
                                  bf16_t* __restrict__ Hcat) {
    const int idx = blockIdx.x * blockDim.x + threadIdx.x;
    if (idx >= BATCH * RH) return;
    const int b = idx >> 7, j = idx & 127;
    float s = 0.0f;
#pragma unroll 4
    for (int r = 0; r < NROUTE; ++r)
        s += Hlast[((long)(b * NROUTE + r)) * RH + j];
    Hcat[(long)b * 192 + j] = (bf16_t)(s * (1.0f / NROUTE));
}

// ---------------------------------------------------------------------------
// Kernel 4: candidate head -> Hcat[:, 128:192]
// ---------------------------------------------------------------------------
__global__ void __launch_bounds__(128) cand_head_kernel(
    const float* __restrict__ cand, const bf16_t* __restrict__ cwp,
    const float* __restrict__ cb, bf16_t* __restrict__ Hcat) {
    const int lane = threadIdx.x & 31;
    const int wg = blockIdx.x * 4 + (threadIdx.x >> 5);  // 32 Mtiles * 4 Ntiles
    const int mt = wg >> 2, nt = wg & 3;
    const int m0 = mt * 16, n0 = nt * 16;
    v8f acc = zero8();
    {
        v16bf a = load_a_f32_full(cand, 2 * FEAT, m0, 0, lane);
        acc = WMMA_BF16(a, load_b_packed(cwp, 4, 0, nt, lane), acc);
    }
    {
        v16bf a = load_a_f32_full(cand, 2 * FEAT, m0, 32, lane);
        acc = WMMA_BF16(a, load_b_packed(cwp, 4, 1, nt, lane), acc);
    }
    {
        v16bf a = load_a_f32_tail72(cand, 2 * FEAT, m0, lane);
        acc = WMMA_BF16(a, load_b_packed(cwp, 4, 2, nt, lane), acc);
    }
    const int cl = lane & 15, hf = (lane >> 4) & 1;
    const int nc = n0 + cl;
    const float bb = cb[nc];
#pragma unroll
    for (int r = 0; r < 8; ++r)
        Hcat[(long)(m0 + r + hf * 8) * 192 + 128 + nc] = (bf16_t)(acc[r] + bb);
}

// ---------------------------------------------------------------------------
// Kernel 5: WMMA GEMM + bias + tanh with packed B:  C[M,N]=tanh(A@B+b)
// ---------------------------------------------------------------------------
__global__ void __launch_bounds__(128) wmma_gemm_bf16_kernel(
    const bf16_t* __restrict__ A, const bf16_t* __restrict__ Bp,
    const float* __restrict__ bias, bf16_t* __restrict__ C,
    int M, int N, int K, int ntiles) {
    const int lane = threadIdx.x & 31;
    const int wg = blockIdx.x * 4 + (threadIdx.x >> 5);
    const int mt = wg / ntiles, nt = wg % ntiles;
    const int m0 = mt * 16, n0 = nt * 16;
    if (m0 >= M) return;
    v8f acc = zero8();
    for (int ki = 0; ki < K / 32; ++ki) {
        v16bf a = load_a_bf16(A, K, m0, ki * 32, lane);
        v16bf b = load_b_packed(Bp, ntiles, ki, nt, lane);
        acc = WMMA_BF16(a, b, acc);
    }
    const int cl = lane & 15, hf = (lane >> 4) & 1;
    const int nc = n0 + cl;
    const float bb = bias[nc];
#pragma unroll
    for (int r = 0; r < 8; ++r)
        C[(long)(m0 + r + hf * 8) * N + nc] = (bf16_t)fast_tanhf(acc[r] + bb);
}

// ---------------------------------------------------------------------------
// Kernel 6: fc3 (N=1) dot product -> d_out[512] f32, vectorized row reads
// ---------------------------------------------------------------------------
__global__ void fc3_kernel(const bf16_t* __restrict__ F2,
                           const float* __restrict__ w,
                           const float* __restrict__ b,
                           float* __restrict__ out) {
    const int i = blockIdx.x * blockDim.x + threadIdx.x;
    if (i >= BATCH) return;
    const uint4* row = (const uint4*)(F2 + (long)i * 128);
    float s = 0.0f;
#pragma unroll
    for (int q = 0; q < 8; ++q) {
        uint4 u = row[q];
        const unsigned uu[4] = {u.x, u.y, u.z, u.w};
#pragma unroll
        for (int d = 0; d < 4; ++d) {
            float lo = __builtin_bit_cast(float, uu[d] << 16);
            float hi = __builtin_bit_cast(float, uu[d] & 0xFFFF0000u);
            s += lo * w[q * 8 + 2 * d] + hi * w[q * 8 + 2 * d + 1];
        }
    }
    out[i] = s + b[0];
}

// ---------------------------------------------------------------------------
extern "C" void kernel_launch(void* const* d_in, const int* in_sizes, int n_in,
                              void* d_out, int out_size, void* d_ws, size_t ws_size,
                              hipStream_t stream) {
    (void)in_sizes; (void)n_in; (void)out_size; (void)ws_size;

    const float* candidates = (const float*)d_in[0];
    const float* customers  = (const float*)d_in[1];
    const float* cust_w1 = (const float*)d_in[2];
    const float* cust_b1 = (const float*)d_in[3];
    const float* cust_w2 = (const float*)d_in[4];
    const float* cust_b2 = (const float*)d_in[5];
    const float* gru_wih0 = (const float*)d_in[6];
    const float* gru_whh0 = (const float*)d_in[7];
    const float* gru_bih0 = (const float*)d_in[8];
    const float* gru_bhh0 = (const float*)d_in[9];
    const float* gru_wih1 = (const float*)d_in[10];
    const float* gru_whh1 = (const float*)d_in[11];
    const float* gru_bih1 = (const float*)d_in[12];
    const float* gru_bhh1 = (const float*)d_in[13];
    const float* cand_w = (const float*)d_in[14];
    const float* cand_b = (const float*)d_in[15];
    const float* fc1_w = (const float*)d_in[16];
    const float* fc1_b = (const float*)d_in[17];
    const float* fc2_w = (const float*)d_in[18];
    const float* fc2_b = (const float*)d_in[19];
    const float* fc3_w = (const float*)d_in[20];
    const float* fc3_b = (const float*)d_in[21];

    char* ws = (char*)d_ws;
    size_t off = 0;
    auto carve = [&](size_t bytes) -> void* {
        void* p = ws + off;
        off += (bytes + 255) & ~(size_t)255;
        return p;
    };
    bf16_t* H2    = (bf16_t*)carve((size_t)NTOK * CO * sizeof(bf16_t));
    float*  H0a   = (float*) carve((size_t)NSEQ * RH * sizeof(float));
    float*  H0b   = (float*) carve((size_t)NSEQ * RH * sizeof(float));
    float*  H1a   = (float*) carve((size_t)NSEQ * RH * sizeof(float));
    float*  H1b   = (float*) carve((size_t)NSEQ * RH * sizeof(float));
    bf16_t* B0a   = (bf16_t*)carve((size_t)NSEQ * RH * sizeof(bf16_t));
    bf16_t* B0b   = (bf16_t*)carve((size_t)NSEQ * RH * sizeof(bf16_t));
    bf16_t* B1a   = (bf16_t*)carve((size_t)NSEQ * RH * sizeof(bf16_t));
    bf16_t* B1b   = (bf16_t*)carve((size_t)NSEQ * RH * sizeof(bf16_t));
    bf16_t* Hcat  = (bf16_t*)carve((size_t)BATCH * 192 * sizeof(bf16_t));
    bf16_t* F1    = (bf16_t*)carve((size_t)BATCH * 128 * sizeof(bf16_t));
    bf16_t* F2b   = (bf16_t*)carve((size_t)BATCH * 128 * sizeof(bf16_t));
    auto carveP = [&](int kt, int ntl) -> bf16_t* {
        return (bf16_t*)carve((size_t)kt * ntl * 512 * sizeof(bf16_t));
    };
    bf16_t* w1p   = carveP(2, 8);
    bf16_t* w2p   = carveP(4, 2);
    bf16_t* wih0p = carveP(1, 24);
    bf16_t* whh0p = carveP(4, 24);
    bf16_t* wih1p = carveP(4, 24);
    bf16_t* whh1p = carveP(4, 24);
    bf16_t* candp = carveP(3, 4);
    bf16_t* fc1p  = carveP(6, 8);
    bf16_t* fc2p  = carveP(4, 8);

    auto pack = [&](const float* B, bf16_t* outp, int K, int N, int kt, int ntl) {
        int total = kt * ntl * 32;
        pack_b_kernel<<<(total + 255) / 256, 256, 0, stream>>>(B, outp, K, N, kt, ntl);
    };
    pack(cust_w1,  w1p,   FEAT, CH, 2, 8);
    pack(cust_w2,  w2p,   CH,   CO, 4, 2);
    pack(gru_wih0, wih0p, CO,   G3, 1, 24);
    pack(gru_whh0, whh0p, RH,   G3, 4, 24);
    pack(gru_wih1, wih1p, RH,   G3, 4, 24);
    pack(gru_whh1, whh1p, RH,   G3, 4, 24);
    pack(cand_w,   candp, 2 * FEAT, 64, 3, 4);
    pack(fc1_w,    fc1p,  192, 128, 6, 8);
    pack(fc2_w,    fc2p,  128, 128, 4, 8);

    (void)hipMemsetAsync(H0a, 0, (size_t)NSEQ * RH * sizeof(float), stream);
    (void)hipMemsetAsync(H1a, 0, (size_t)NSEQ * RH * sizeof(float), stream);
    (void)hipMemsetAsync(B0a, 0, (size_t)NSEQ * RH * sizeof(bf16_t), stream);
    (void)hipMemsetAsync(B1a, 0, (size_t)NSEQ * RH * sizeof(bf16_t), stream);

    // 1) customer MLP
    customer_mlp_kernel<<<NTOK / 16 / 4, 128, 0, stream>>>(
        customers, w1p, cust_b1, w2p, cust_b2, H2);

    // 2) GRU scan, both layers per timestep
    const int gru_blocks = (NSEQ / 32) * 8 / 4;  // 1536
    float  *h0i = H0a, *h0o = H0b, *h1i = H1a, *h1o = H1b;
    bf16_t *b0i = B0a, *b0o = B0b, *b1i = B1a, *b1o = B1b;
    for (int t = 0; t < NNODE; ++t) {
        gru_step_kernel<CO><<<gru_blocks, 128, 0, stream>>>(
            h0i, h0o, b0i, b0o, H2 + (size_t)t * NSEQ * CO,
            wih0p, whh0p, gru_bih0, gru_bhh0);
        gru_step_kernel<RH><<<gru_blocks, 128, 0, stream>>>(
            h1i, h1o, b1i, b1o, b0o,
            wih1p, whh1p, gru_bih1, gru_bhh1);
        float*  tf; bf16_t* tb;
        tf = h0i; h0i = h0o; h0o = tf;   tb = b0i; b0i = b0o; b0o = tb;
        tf = h1i; h1i = h1o; h1o = tf;   tb = b1i; b1i = b1o; b1o = tb;
    }

    // 3) route mean + candidate head -> Hcat[512,192]
    route_mean_kernel<<<(BATCH * RH + 255) / 256, 256, 0, stream>>>(h1i, Hcat);
    cand_head_kernel<<<32, 128, 0, stream>>>(candidates, candp, cand_b, Hcat);

    // 4) fc1 (K=192), fc2 (K=128), tanh
    wmma_gemm_bf16_kernel<<<(32 * 8) / 4, 128, 0, stream>>>(
        Hcat, fc1p, fc1_b, F1, BATCH, 128, 192, 8);
    wmma_gemm_bf16_kernel<<<(32 * 8) / 4, 128, 0, stream>>>(
        F1, fc2p, fc2_b, F2b, BATCH, 128, 128, 8);

    // 5) fc3 -> d_out[512]
    fc3_kernel<<<(BATCH + 255) / 256, 256, 0, stream>>>(
        F2b, fc3_w, fc3_b, (float*)d_out);
}
